// LurieNet_k_75067438399570
// MI455X (gfx1250) — matrix operand order repaired and checked by reading, not compiled, not run on hardware
//
#include <hip/hip_runtime.h>
#include <math.h>

#define NDIM 128
#define TSTEPS 512
#define BSZ 512
#define XS 136            // padded LDS row stride (halves) to spread banks

typedef __attribute__((ext_vector_type(16))) _Float16 v16h;
typedef __attribute__((ext_vector_type(8)))  float    v8f;

union V16 { uint4 q[2]; v16h v; };

// ---------------------------------------------------------------------------
// Setup kernels (128x128 fp32, one-time cost; latency-irrelevant)
// ---------------------------------------------------------------------------

__global__ void k_skew_scale(const float* __restrict__ Z, float* __restrict__ M, float scale) {
  int idx = blockIdx.x * blockDim.x + threadIdx.x;
  int i = idx >> 7, j = idx & 127;
  float u = (j > i) ? Z[i * NDIM + j] : 0.0f;
  float l = (i > j) ? Z[j * NDIM + i] : 0.0f;
  M[idx] = (u - l) * scale;
}

// R = I + T * inv
__global__ void k_axpyI(float* __restrict__ R, const float* __restrict__ T, float inv) {
  int idx = blockIdx.x * blockDim.x + threadIdx.x;
  int i = idx >> 7, j = idx & 127;
  R[idx] = ((i == j) ? 1.0f : 0.0f) + T[idx] * inv;
}

// C = A @ B  (128x128x128)
__global__ void k_gemm(float* __restrict__ C, const float* __restrict__ A,
                       const float* __restrict__ B) {
  int idx = blockIdx.x * blockDim.x + threadIdx.x;
  int i = idx >> 7, j = idx & 127;
  float s = 0.0f;
  for (int k = 0; k < NDIM; ++k) s = fmaf(A[i * NDIM + k], B[k * NDIM + j], s);
  C[idx] = s;
}

__global__ void k_copy(float* __restrict__ D, const float* __restrict__ S) {
  int idx = blockIdx.x * blockDim.x + threadIdx.x;
  D[idx] = S[idx];
}

// alpha = sqrt(4*(G^2/K) * sum_k topk(|diag B|)^2 * topk(|diag C|)^2),  G=1, K=4
__global__ void k_alpha(const float* __restrict__ ZB, const float* __restrict__ ZC,
                        float* __restrict__ out) {
  float tb[4] = {0.f, 0.f, 0.f, 0.f}, tc[4] = {0.f, 0.f, 0.f, 0.f};
  for (int i = 0; i < NDIM; ++i) {
    float vb = fabsf(ZB[i * (NDIM + 1)]);
    if (vb > tb[3]) {
      tb[3] = vb;
      for (int q = 3; q > 0; --q)
        if (tb[q] > tb[q - 1]) { float t = tb[q]; tb[q] = tb[q - 1]; tb[q - 1] = t; }
    }
    float vc = fabsf(ZC[i * (NDIM + 1)]);
    if (vc > tc[3]) {
      tc[3] = vc;
      for (int q = 3; q > 0; --q)
        if (tc[q] > tc[q - 1]) { float t = tc[q]; tc[q] = tc[q - 1]; tc[q - 1] = t; }
    }
  }
  float s = 0.0f;
  for (int q = 0; q < 4; ++q) s += tb[q] * tb[q] * tc[q] * tc[q];
  out[0] = sqrtf(4.0f * (1.0f / 4.0f) * s);
}

// W[i][j] = |Zdiag[i][i]| * V[j][i]   (W = diag(s) @ V^T)
__global__ void k_diagscaleT(float* __restrict__ W, const float* __restrict__ V,
                             const float* __restrict__ Zdiag) {
  int idx = blockIdx.x * blockDim.x + threadIdx.x;
  int i = idx >> 7, j = idx & 127;
  W[idx] = fabsf(Zdiag[i * (NDIM + 1)]) * V[j * NDIM + i];
}

// W[i][j] = -(alpha + |ZG[i][i]|) * U[j][i]
__global__ void k_diagscaleT_A(float* __restrict__ W, const float* __restrict__ U,
                               const float* __restrict__ ZG, const float* __restrict__ alpha) {
  int idx = blockIdx.x * blockDim.x + threadIdx.x;
  int i = idx >> 7, j = idx & 127;
  W[idx] = -(alpha[0] + fabsf(ZG[i * (NDIM + 1)])) * U[j * NDIM + i];
}

// A = 0.5*P + 0.5*skew(ZY)
__global__ void k_finalizeA(float* __restrict__ A, const float* __restrict__ P,
                            const float* __restrict__ ZY) {
  int idx = blockIdx.x * blockDim.x + threadIdx.x;
  int i = idx >> 7, j = idx & 127;
  float u = (j > i) ? ZY[i * NDIM + j] : 0.0f;
  float l = (i > j) ? ZY[j * NDIM + i] : 0.0f;
  A[idx] = 0.5f * P[idx] + 0.5f * (u - l);
}

__global__ void k_tof16(_Float16* __restrict__ D, const float* __restrict__ S) {
  int idx = blockIdx.x * blockDim.x + threadIdx.x;
  D[idx] = (_Float16)S[idx];
}

// ---------------------------------------------------------------------------
// Main recurrence kernel: 32 blocks (batch slices of 16) x 256 threads (8 waves)
// Wave w owns rows [16w, 16w+16); weights live in registers in WMMA A-layout.
// State x is exchanged through LDS as f16, column-major with padded stride.
// ---------------------------------------------------------------------------

// pack 2 floats -> packed f16 pair with one v_cvt_pk_rtz_f16_f32
static __device__ __forceinline__ unsigned pk2(float a, float b) {
  typedef __attribute__((ext_vector_type(2))) __fp16 fp16x2;
  union { fp16x2 h; unsigned u; } z;
  z.h = __builtin_amdgcn_cvt_pkrtz(a, b);
  return z.u;
}

static __device__ __forceinline__ float ftanh(float x) {
#if __has_builtin(__builtin_amdgcn_tanhf)
  return __builtin_amdgcn_tanhf(x);          // v_tanh_f32
#else
  float e = __expf(2.0f * x);                // v_exp_f32
  float r = __builtin_amdgcn_rcpf(e + 1.0f); // v_rcp_f32
  return fmaf(-2.0f, r, 1.0f);
#endif
}

// Workgroup barrier that orders only LDS traffic (DScnt), NOT global stores.
// __syncthreads() waits s_wait_storecnt 0x0 as well, which would serialize the
// per-step global output stores (L2 ack latency) into the recurrence's
// critical path. The output stores are never read, so that fence is dead.
static __device__ __forceinline__ void lds_barrier() {
  asm volatile(
      "s_wait_dscnt 0x0\n\t"
      "s_barrier_signal -1\n\t"
      "s_barrier_wait -1" ::: "memory");
}

__global__ __launch_bounds__(256) void k_lurie(
    const float* __restrict__ X0, const _Float16* __restrict__ Ag,
    const _Float16* __restrict__ Bg, const _Float16* __restrict__ Cg,
    const float* __restrict__ bx, const float* __restrict__ by,
    float* __restrict__ out) {
  __shared__ __align__(16) _Float16 xsh[2 * 16 * XS];  // double-buffered state, col-major
  __shared__ __align__(16) _Float16 tsh[16 * XS];      // tanh(y) buffer

  const int tid = threadIdx.x;
  const int w = tid >> 5;     // wave id = row tile
  const int lane = tid & 31;
  const int col = lane & 15;  // batch column within slice
  const int hi = lane >> 4;   // half-wave select
  const int b = (blockIdx.x << 4) + col;        // global batch index
  const int n0 = (w << 4) + (hi << 3);          // first row held (C/D layout)

  // ---- load per-wave weight slabs into registers (WMMA A-operand layout) ----
  V16 Aw[4], Bw[4], Cw[4];
  {
    const int row = (w << 4) + col;  // A-layout: M = lane % 16
    const int ks = hi << 3;          // half-wave K offset
#pragma unroll
    for (int c = 0; c < 4; ++c) {
      const _Float16* pa = Ag + row * NDIM + 32 * c + ks;
      Aw[c].q[0] = *(const uint4*)pa; Aw[c].q[1] = *(const uint4*)(pa + 16);
      const _Float16* pb = Bg + row * NDIM + 32 * c + ks;
      Bw[c].q[0] = *(const uint4*)pb; Bw[c].q[1] = *(const uint4*)(pb + 16);
      const _Float16* pc = Cg + row * NDIM + 32 * c + ks;
      Cw[c].q[0] = *(const uint4*)pc; Cw[c].q[1] = *(const uint4*)(pc + 16);
    }
  }

  // ---- biases in C/D layout ----
  float bxv[8], byv[8];
  {
    float4 t0 = *(const float4*)(bx + n0), t1 = *(const float4*)(bx + n0 + 4);
    bxv[0] = t0.x; bxv[1] = t0.y; bxv[2] = t0.z; bxv[3] = t0.w;
    bxv[4] = t1.x; bxv[5] = t1.y; bxv[6] = t1.z; bxv[7] = t1.w;
    float4 u0 = *(const float4*)(by + n0), u1 = *(const float4*)(by + n0 + 4);
    byv[0] = u0.x; byv[1] = u0.y; byv[2] = u0.z; byv[3] = u0.w;
    byv[4] = u1.x; byv[5] = u1.y; byv[6] = u1.z; byv[7] = u1.w;
  }

  // ---- initial state: load X0, emit t=0 output, seed LDS buffer 0 ----
  float xr[8];
  {
    const float* px = X0 + (size_t)b * NDIM + n0;
    float4 a0 = *(const float4*)px, a1 = *(const float4*)(px + 4);
    xr[0] = a0.x; xr[1] = a0.y; xr[2] = a0.z; xr[3] = a0.w;
    xr[4] = a1.x; xr[5] = a1.y; xr[6] = a1.z; xr[7] = a1.w;
    float4* po = (float4*)(out + (size_t)b * TSTEPS * NDIM + n0);
    po[0] = a0; po[1] = a1;
    uint4 pk;
    pk.x = pk2(xr[0], xr[1]); pk.y = pk2(xr[2], xr[3]);
    pk.z = pk2(xr[4], xr[5]); pk.w = pk2(xr[6], xr[7]);
    *(uint4*)(xsh + col * XS + n0) = pk;
  }
  lds_barrier();

  for (int t = 1; t < TSTEPS; ++t) {
    const _Float16* curx = xsh + ((t - 1) & 1) * (16 * XS);
    _Float16* nxtx = xsh + (t & 1) * (16 * XS);

    // ---- preload all 4 x-operand chunks (reused by C@x and A@x) ----
    V16 xop[4];
#pragma unroll
    for (int c = 0; c < 4; ++c) {
      const _Float16* p = curx + col * XS + 32 * c + (hi << 3);
      xop[c].q[0] = *(const uint4*)p;
      xop[c].q[1] = *(const uint4*)(p + 16);
    }

    // ---- phase 1: y = C @ x + by ; tanh ; share ----
    v8f acc;
#pragma unroll
    for (int r = 0; r < 8; ++r) acc[r] = byv[r];
#pragma unroll
    for (int c = 0; c < 4; ++c)
      acc = __builtin_amdgcn_wmma_f32_16x16x32_f16(false, Cw[c].v, false, xop[c].v,
                                                   (short)0, acc, false, false);
    {
      uint4 pk;
      pk.x = pk2(ftanh(acc[0]), ftanh(acc[1]));
      pk.y = pk2(ftanh(acc[2]), ftanh(acc[3]));
      pk.z = pk2(ftanh(acc[4]), ftanh(acc[5]));
      pk.w = pk2(ftanh(acc[6]), ftanh(acc[7]));
      *(uint4*)(tsh + col * XS + n0) = pk;
    }
    lds_barrier();

    // ---- phase 2: xn = x + dt*(A@x + B@tanh + bx) ----
    // preload tanh operands; A@x WMMAs below run on registers while these fly
    V16 top[4];
#pragma unroll
    for (int c = 0; c < 4; ++c) {
      const _Float16* p = tsh + col * XS + 32 * c + (hi << 3);
      top[c].q[0] = *(const uint4*)p;
      top[c].q[1] = *(const uint4*)(p + 16);
    }

    v8f a2;
#pragma unroll
    for (int r = 0; r < 8; ++r) a2[r] = bxv[r];
#pragma unroll
    for (int c = 0; c < 4; ++c)
      a2 = __builtin_amdgcn_wmma_f32_16x16x32_f16(false, Aw[c].v, false, xop[c].v,
                                                  (short)0, a2, false, false);
#pragma unroll
    for (int c = 0; c < 4; ++c)
      a2 = __builtin_amdgcn_wmma_f32_16x16x32_f16(false, Bw[c].v, false, top[c].v,
                                                  (short)0, a2, false, false);
#pragma unroll
    for (int r = 0; r < 8; ++r) xr[r] = fmaf(0.01f, a2[r], xr[r]);

    {
      uint4 pk;
      pk.x = pk2(xr[0], xr[1]); pk.y = pk2(xr[2], xr[3]);
      pk.z = pk2(xr[4], xr[5]); pk.w = pk2(xr[6], xr[7]);
      *(uint4*)(nxtx + col * XS + n0) = pk;
    }
    lds_barrier();

    float4 o0, o1;
    o0.x = xr[0]; o0.y = xr[1]; o0.z = xr[2]; o0.w = xr[3];
    o1.x = xr[4]; o1.y = xr[5]; o1.z = xr[6]; o1.w = xr[7];
    float4* po = (float4*)(out + ((size_t)b * TSTEPS + t) * NDIM + n0);
    po[0] = o0; po[1] = o1;
  }
}

// ---------------------------------------------------------------------------
// Host orchestration
// ---------------------------------------------------------------------------

static void expm_skew(const float* Z, float* dest, float* M, float* R, float* T,
                      hipStream_t st) {
  const int GB = 64, TB = 256;
  // scale by 1/2^8, Taylor q=8 via Horner, then 8 squarings
  k_skew_scale<<<GB, TB, 0, st>>>(Z, M, 1.0f / 256.0f);
  k_axpyI<<<GB, TB, 0, st>>>(R, M, 1.0f / 8.0f);  // R = I + M/8
  for (int k = 7; k >= 1; --k) {
    k_gemm<<<GB, TB, 0, st>>>(T, M, R);
    k_axpyI<<<GB, TB, 0, st>>>(R, T, 1.0f / (float)k);  // R = I + (M@R)/k
  }
  float* cur = R; float* oth = T;
  for (int i = 0; i < 8; ++i) {
    k_gemm<<<GB, TB, 0, st>>>(oth, cur, cur);
    float* tmp = cur; cur = oth; oth = tmp;
  }
  k_copy<<<GB, TB, 0, st>>>(dest, cur);
}

extern "C" void kernel_launch(void* const* d_in, const int* in_sizes, int n_in,
                              void* d_out, int out_size, void* d_ws, size_t ws_size,
                              hipStream_t stream) {
  const float* X0   = (const float*)d_in[0];
  const float* ZA_Y = (const float*)d_in[1];
  const float* ZA_U = (const float*)d_in[2];
  const float* ZA_G = (const float*)d_in[3];
  const float* ZB_U = (const float*)d_in[4];
  const float* ZB_V = (const float*)d_in[5];
  const float* ZB_S = (const float*)d_in[6];
  const float* ZC_U = (const float*)d_in[7];
  const float* ZC_V = (const float*)d_in[8];
  const float* ZC_S = (const float*)d_in[9];
  const float* bx   = (const float*)d_in[10];
  const float* by   = (const float*)d_in[11];
  float* out = (float*)d_out;

  float* ws = (float*)d_ws;
  const size_t M = (size_t)NDIM * NDIM;
  float* TMP = ws + 0 * M;
  float* Rb  = ws + 1 * M;
  float* Tb  = ws + 2 * M;
  float* UA  = ws + 3 * M;
  float* UB  = ws + 4 * M;
  float* VB  = ws + 5 * M;
  float* UC  = ws + 6 * M;
  float* VC  = ws + 7 * M;
  float* Wb  = ws + 8 * M;
  float* Pb  = ws + 9 * M;
  _Float16* Af = (_Float16*)(ws + 10 * M);
  _Float16* Bf = Af + M;
  _Float16* Cf = Bf + M;
  float* alphaBuf = (float*)(Cf + M);

  const int GB = 64, TB = 256;

  // orthogonal factors via expm of skew matrices
  expm_skew(ZA_U, UA, TMP, Rb, Tb, stream);
  expm_skew(ZB_U, UB, TMP, Rb, Tb, stream);
  expm_skew(ZB_V, VB, TMP, Rb, Tb, stream);
  expm_skew(ZC_U, UC, TMP, Rb, Tb, stream);
  expm_skew(ZC_V, VC, TMP, Rb, Tb, stream);

  // alpha from top-4 singular values of diag parametrizations
  k_alpha<<<1, 1, 0, stream>>>(ZB_S, ZC_S, alphaBuf);

  // C = UC @ diag(|ZC_S|) @ VC^T
  k_diagscaleT<<<GB, TB, 0, stream>>>(Wb, VC, ZC_S);
  k_gemm<<<GB, TB, 0, stream>>>(Pb, UC, Wb);
  k_tof16<<<GB, TB, 0, stream>>>(Cf, Pb);

  // B = UB @ diag(|ZB_S|) @ VB^T
  k_diagscaleT<<<GB, TB, 0, stream>>>(Wb, VB, ZB_S);
  k_gemm<<<GB, TB, 0, stream>>>(Pb, UB, Wb);
  k_tof16<<<GB, TB, 0, stream>>>(Bf, Pb);

  // A = 0.5 * UA @ diag(-(alpha+|ZA_G|)) @ UA^T + 0.5 * skew(ZA_Y)
  k_diagscaleT_A<<<GB, TB, 0, stream>>>(Wb, UA, ZA_G, alphaBuf);
  k_gemm<<<GB, TB, 0, stream>>>(Pb, UA, Wb);
  k_finalizeA<<<GB, TB, 0, stream>>>(Tb, Pb, ZA_Y);
  k_tof16<<<GB, TB, 0, stream>>>(Af, Tb);

  // main recurrence: 32 batch slices x 8 waves
  k_lurie<<<32, 256, 0, stream>>>(X0, Af, Bf, Cf, bx, by, out);
}
